// Experts_3719441678634
// MI455X (gfx1250) — matrix operand order repaired and checked by reading, not compiled, not run on hardware
//
#include <hip/hip_runtime.h>
#include <math.h>

// ---------- CDNA5 WMMA types ----------
typedef __attribute__((ext_vector_type(16))) __bf16 v16bf;
typedef __attribute__((ext_vector_type(2)))  __bf16 v2bf;
typedef __attribute__((ext_vector_type(2)))  float  v2f;
typedef __attribute__((ext_vector_type(8)))  float  v8f;
typedef __attribute__((ext_vector_type(8)))  unsigned int v8u;

// ---------- problem constants ----------
static constexpr int Bsz  = 2;
static constexpr int Esz  = 8;
static constexpr int Nseq = 1024;
static constexpr int Dsz  = 1024;
static constexpr int Fsz  = 4096;
static constexpr int ROWS = Bsz * Esz * Nseq;   // 16384 rows, grouped by (b,e)

// ---------- tile config ----------
static constexpr int BM  = 128;   // rows per workgroup tile
static constexpr int BN  = 128;   // cols per workgroup tile
static constexpr int BK  = 32;    // K per step (one bf16 WMMA K)
static constexpr int LDH = 40;    // LDS row stride in halves (80B, 16B aligned, padded)

// packed fp32x2 -> bf16x2 (lowers to v_cvt_pk_bf16_f32)
__device__ __forceinline__ unsigned int pkbf(float a, float b) {
  v2f  f = {a, b};
  v2bf r = __builtin_convertvector(f, v2bf);
  return __builtin_bit_cast(unsigned int, r);
}
__device__ __forceinline__ unsigned short f2bf(float f) {
  return __builtin_bit_cast(unsigned short, (__bf16)f);
}
__device__ __forceinline__ float gelu_exact(float x) {
  return 0.5f * x * (1.0f + erff(x * 0.70710678118654752440f));
}

__device__ __forceinline__ v16bf ld_frag(const unsigned short* p, int offLo, int offHi) {
  const uint4 lo = *(const uint4*)(p + offLo);   // 8 halves
  const uint4 hi = *(const uint4*)(p + offHi);   // 8 halves
  v8u u = (v8u){lo.x, lo.y, lo.z, lo.w, hi.x, hi.y, hi.z, hi.w};
  return __builtin_bit_cast(v16bf, u);
}

// One kernel covers both GEMMs:
//  A_BF16=false, APPLY_GELU=true : h_bf16 = gelu(x_f32 @ W1_e + b1_e)   (write Obf)
//  A_BF16=true,  APPLY_GELU=false: out_f32 = h_bf16 @ W2_e + b2_e       (write Of)
//  A_BF16 path copies the A tile with global_load_async_to_lds_b128 (ASYNCcnt).
template <bool A_BF16, bool APPLY_GELU>
__global__ __launch_bounds__(256)
void ffn_gemm(const float* __restrict__ Af,            // fp32 A (x), when !A_BF16
              const unsigned short* __restrict__ Ab,   // bf16 A (h), when A_BF16
              const float* __restrict__ Wt,            // [E][K][Nw] fp32 weights
              const float* __restrict__ bias,          // [E][Nw]
              unsigned short* __restrict__ Obf,        // bf16 out (h), stride Nw
              float* __restrict__ Of,                  // fp32 out,     stride Nw
              int K, int Nw)
{
  __shared__ alignas(16) unsigned short sA[2][BM * LDH];  // A tile, [m][k]
  __shared__ alignas(16) unsigned short sB[2][BM * LDH];  // B tile, [n][k] (K-contig per col)

  const int tid  = threadIdx.x;
  const int lane = tid & 31;
  const int wave = tid >> 5;
  const int wm   = wave >> 1;      // 0..3 : 32-row slab within tile
  const int wn   = wave & 1;       // 0..1 : 64-col slab within tile

  const int row0 = blockIdx.y * BM;
  const int n0   = blockIdx.x * BN;
  const int e    = (row0 / Nseq) % Esz;      // BM=128 divides Nseq=1024 -> no straddle
  const float* Wp = Wt   + (size_t)e * K * Nw;
  const float* bp = bias + (size_t)e * Nw;

  // staging coordinates
  const int ar  = tid >> 3;        const int ac  = (tid & 7) * 4;  // fp32 A: 4 iters, +32 rows
  const int a2r = tid >> 2;        const int a2c = (tid & 3) * 8;  // bf16 A: 2 iters, +64 rows
  const int bkb = (tid >> 5) * 4;  const int bnb = (tid & 31) * 4; // B: one 4x4 micro-block

  float4 aS[4]; float4 bS[4];

  // ---- A: fp32 global -> regs -> packed-cvt -> LDS (GEMM1) ----
  auto gloadA = [&](int kt) {
    const int k0 = kt * BK;
    #pragma unroll
    for (int i = 0; i < 4; ++i)
      aS[i] = *(const float4*)(Af + (size_t)(row0 + ar + 32 * i) * K + k0 + ac);
  };
  auto sstoreA = [&](int buf) {
    unsigned short* pa = sA[buf];
    #pragma unroll
    for (int i = 0; i < 4; ++i) {
      uint2 v;
      v.x = pkbf(aS[i].x, aS[i].y);
      v.y = pkbf(aS[i].z, aS[i].w);
      *(uint2*)(pa + (ar + 32 * i) * LDH + ac) = v;
    }
  };

  // ---- A: bf16 global -> LDS via CDNA5 async copy (GEMM2), no VGPR staging ----
  auto asyncA = [&](int kt, int buf) {
    const int k0 = kt * BK;
    #pragma unroll
    for (int i = 0; i < 2; ++i) {
      const unsigned short* g = Ab + (size_t)(row0 + a2r + 64 * i) * K + k0 + a2c;
      unsigned lds = (unsigned)(size_t)(const void*)&sA[buf][(a2r + 64 * i) * LDH + a2c];
      asm volatile("global_load_async_to_lds_b128 %0, %1, off"
                   :: "v"(lds), "v"(g) : "memory");
    }
  };
  auto waitAsync = [&]() {
    asm volatile("s_wait_asynccnt 0x0" ::: "memory");
  };

  // ---- B: fp32 4x4 micro-block -> register transpose -> packed b64 LDS stores ----
  auto gloadB = [&](int kt) {
    const int k0 = kt * BK;
    #pragma unroll
    for (int j = 0; j < 4; ++j)
      bS[j] = *(const float4*)(Wp + (size_t)(k0 + bkb + j) * Nw + n0 + bnb);
  };
  auto sstoreB = [&](int buf) {
    unsigned short* pb = sB[buf];
    const float c[4][4] = {{bS[0].x, bS[1].x, bS[2].x, bS[3].x},
                           {bS[0].y, bS[1].y, bS[2].y, bS[3].y},
                           {bS[0].z, bS[1].z, bS[2].z, bS[3].z},
                           {bS[0].w, bS[1].w, bS[2].w, bS[3].w}};
    #pragma unroll
    for (int i = 0; i < 4; ++i) {
      uint2 v;
      v.x = pkbf(c[i][0], c[i][1]);
      v.y = pkbf(c[i][2], c[i][3]);
      *(uint2*)(pb + (bnb + i) * LDH + bkb) = v;   // [n][k], 8B store
    }
  };

  v8f acc[2][4] = {};  // fp32 accumulators (8 VGPRs each)

  const int KT = K / BK;

  // prologue: fill buffer 0
  if (A_BF16) asyncA(0, 0); else { gloadA(0); }
  gloadB(0);
  if (!A_BF16) sstoreA(0);
  sstoreB(0);
  if (A_BF16) waitAsync();
  __syncthreads();

  const int l15 = lane & 15;

  for (int kt = 0; kt < KT; ++kt) {
    const int buf = kt & 1;

    if (kt + 1 < KT) {                        // overlap next-tile movement with math
      if (A_BF16) asyncA(kt + 1, buf ^ 1);    // DMA straight into LDS
      else        gloadA(kt + 1);
      gloadB(kt + 1);
    }
    if (kt + 2 < KT) {                        // global_prefetch_b8, 2 steps ahead
      if (!A_BF16)
        __builtin_prefetch(Af + (size_t)(row0 + ar) * K + (kt + 2) * BK + ac, 0, 1);
      else
        __builtin_prefetch(Ab + (size_t)(row0 + a2r) * K + (kt + 2) * BK + a2c, 0, 1);
      __builtin_prefetch(Wp + (size_t)((kt + 2) * BK + bkb) * Nw + n0 + bnb, 0, 1);
    }

    const unsigned short* pa = sA[buf];
    const unsigned short* pb = sB[buf];

    // A fragment per ISA 16-bit A layout:
    //   lanes 0-15 : halves[0..7]=K0..7,  halves[8..15]=K16..23
    //   lanes16-31 : halves[0..7]=K8..15, halves[8..15]=K24..31
    v16bf afr[2], bfr[4];
    const int kbA = (lane < 16) ? 0 : 8;
    #pragma unroll
    for (int mi = 0; mi < 2; ++mi) {
      const int off = (wm * 32 + mi * 16 + l15) * LDH + kbA;
      afr[mi] = ld_frag(pa, off, off + 16);
    }
    // B fragment: lane holds one column; lanes 0-15 -> K0..7,K16..23, lanes 16-31 -> K8..15,K24..31
    const int kbB = (lane < 16) ? 0 : 8;
    #pragma unroll
    for (int ni = 0; ni < 4; ++ni) {
      const int off = (wn * 64 + ni * 16 + l15) * LDH + kbB;
      bfr[ni] = ld_frag(pb, off, off + 16);
    }

    #pragma unroll
    for (int mi = 0; mi < 2; ++mi)
      #pragma unroll
      for (int ni = 0; ni < 4; ++ni)
        acc[mi][ni] = __builtin_amdgcn_wmma_f32_16x16x32_bf16(
            false, afr[mi], false, bfr[ni], (short)0, acc[mi][ni], false, false);

    if (kt + 1 < KT) {
      if (!A_BF16) sstoreA(buf ^ 1);
      sstoreB(buf ^ 1);
      if (A_BF16) waitAsync();               // async A tile landed in LDS
    }
    __syncthreads();
  }

  // epilogue: C layout -> VGPR j: lanes 0-15 are M=j, lanes 16-31 are M=8+j
  const int rsel = (lane >> 4) * 8;
  #pragma unroll
  for (int mi = 0; mi < 2; ++mi) {
    #pragma unroll
    for (int ni = 0; ni < 4; ++ni) {
      const int gc = n0 + wn * 64 + ni * 16 + l15;
      const float bb = bp[gc];
      #pragma unroll
      for (int j = 0; j < 8; ++j) {
        const int gr = row0 + wm * 32 + mi * 16 + rsel + j;
        float v = acc[mi][ni][j] + bb;
        if (APPLY_GELU) {
          v = gelu_exact(v);
          Obf[(size_t)gr * Nw + gc] = f2bf(v);
        } else {
          Of[(size_t)gr * Nw + gc] = v;
        }
      }
    }
  }
}

extern "C" void kernel_launch(void* const* d_in, const int* in_sizes, int n_in,
                              void* d_out, int out_size, void* d_ws, size_t ws_size,
                              hipStream_t stream) {
  const float* x  = (const float*)d_in[0];   // [B,E,N,D]
  const float* w1 = (const float*)d_in[1];   // [E,D,F]
  const float* b1 = (const float*)d_in[2];   // [E,F]
  const float* w2 = (const float*)d_in[3];   // [E,F,D]
  const float* b2 = (const float*)d_in[4];   // [E,D]
  float* out = (float*)d_out;                // [B,E,N,D]
  unsigned short* h = (unsigned short*)d_ws; // bf16 [ROWS, F] intermediate (128 MB)

  dim3 blk(256);

  // GEMM1: h = gelu(x @ w1 + b1)   (M=16384, K=1024, N=4096)
  dim3 g1(Fsz / BN, ROWS / BM);
  ffn_gemm<false, true><<<g1, blk, 0, stream>>>(x, nullptr, w1, b1, h, nullptr, Dsz, Fsz);

  // GEMM2: out = h @ w2 + b2       (M=16384, K=4096, N=1024)
  dim3 g2(Dsz / BN, ROWS / BM);
  ffn_gemm<true, false><<<g2, blk, 0, stream>>>(nullptr, h, w2, b2, nullptr, out, Fsz, Dsz);
}
